// CrossAttentionBlock_24180665876962
// MI455X (gfx1250) — matrix operand order repaired and checked by reading, not compiled, not run on hardware
//
#include <hip/hip_runtime.h>

#define B_   4
#define C_   512
#define C8_  64
#define N_   4096
#define EPS_ 1e-5f

typedef __attribute__((ext_vector_type(16))) __bf16        v16bf;
typedef __attribute__((ext_vector_type(8)))  float         v8f;
typedef __attribute__((ext_vector_type(8)))  unsigned short v8us;
typedef __attribute__((ext_vector_type(16))) unsigned short v16us;

static __device__ __forceinline__ unsigned short f2bf(float f) {
    unsigned int u = __float_as_uint(f);
    u = (u + 0x7FFFu + ((u >> 16) & 1u)) >> 16;   // round-to-nearest-even
    return (unsigned short)u;
}

static __device__ __forceinline__ v8f wmma_bf16(v16us a, v16us b, v8f c) {
    return __builtin_amdgcn_wmma_f32_16x16x32_bf16(
        false, __builtin_bit_cast(v16bf, a),
        false, __builtin_bit_cast(v16bf, b),
        (short)0, c, false, false);
}

static __device__ __forceinline__ unsigned lds_off(const void* p) {
    // generic (flat) pointers to LDS carry the LDS byte offset in addr[31:0]
    return (unsigned)(unsigned long long)p;
}

static __device__ __forceinline__ void async_load_b128(unsigned lds, const void* gptr) {
    asm volatile("global_load_async_to_lds_b128 %0, %1, off"
                 :: "v"(lds), "v"(gptr) : "memory");
}

// ---------------------------------------------------------------------------
// Kernel 1: three 1x1-conv projections, output bf16 in WMMA-friendly layouts.
// grid = (N/256, 3, B); proj & batch uniform per block -> scalar weight loads.
//   pj=0: g     (x0) -> gT     [B][64][N]   (B-side, K=m contiguous)
//   pj=1: theta (x1) -> thetaT [B][N][64]   (B-side, K=c contiguous)
//   pj=2: phi   (x1) -> phi    [B][N][64]   (A-side rows)
// ---------------------------------------------------------------------------
__global__ void __launch_bounds__(256) proj_kernel(
    const float* __restrict__ x0, const float* __restrict__ x1,
    const float* __restrict__ g_w, const float* __restrict__ g_b,
    const float* __restrict__ th_w, const float* __restrict__ th_b,
    const float* __restrict__ ph_w, const float* __restrict__ ph_b,
    unsigned short* __restrict__ phi, unsigned short* __restrict__ thetaT,
    unsigned short* __restrict__ gT)
{
    const int n  = blockIdx.x * 256 + threadIdx.x;
    const int pj = blockIdx.y;
    const int b  = blockIdx.z;
    const float* x    = (pj == 0) ? x0  : x1;
    const float* w    = (pj == 0) ? g_w : (pj == 1) ? th_w : ph_w;
    const float* bias = (pj == 0) ? g_b : (pj == 1) ? th_b : ph_b;
    const float* xs   = x + (size_t)b * C_ * N_ + n;

    float acc[64];
#pragma unroll
    for (int o = 0; o < 64; ++o) acc[o] = 0.f;
    for (int c = 0; c < C_; ++c) {
        float xv = xs[(size_t)c * N_];
#pragma unroll
        for (int o = 0; o < 64; ++o)
            acc[o] = fmaf(w[o * C_ + c], xv, acc[o]);
    }
    if (pj == 0) {
        unsigned short* d = gT + (size_t)b * C8_ * N_ + n;
#pragma unroll
        for (int o = 0; o < 64; ++o) d[(size_t)o * N_] = f2bf(acc[o] + bias[o]);
    } else {
        unsigned short* d = ((pj == 1) ? thetaT : phi) + ((size_t)b * N_ + n) * C8_;
#pragma unroll
        for (int o = 0; o < 64; ++o) d[o] = f2bf(acc[o] + bias[o]);
    }
}

// ---------------------------------------------------------------------------
// Kernel 2: flash attention, one wave per 16-row tile, 8 waves per block.
// b and the m0 loop are block-uniform -> the whole block cooperatively stages
// each theta (32x64) and g (64x32) bf16 chunk into LDS with
// global_load_async_to_lds_b128 (ASYNCcnt), double-buffered so copy of chunk
// i+1 overlaps compute of chunk i.  Per chunk: 4 score WMMAs + per-wave
// online softmax through an LDS tile + 4 y-accumulation WMMAs.
// ---------------------------------------------------------------------------
#define SROW 34   // Sbuf f32 row stride (bank spread)
#define PROW 40   // Pbuf bf16 row stride (mult of 8 -> 16B-aligned ds b128)
#define KROW 72   // Kst  bf16 row stride (mult of 8, 36-dword bank rotation)
#define GROW 40   // Gst  bf16 row stride (mult of 8, 20-dword bank rotation)

__global__ void __launch_bounds__(256) attn_kernel(
    const unsigned short* __restrict__ phi,     // [B][N][64]
    const unsigned short* __restrict__ thetaT,  // [B][N][64]
    const unsigned short* __restrict__ gT,      // [B][64][N]
    float* __restrict__ yflat)                  // [B][N*64]
{
    __shared__ alignas(16) float          Sbuf[8 * 16 * SROW];
    __shared__ alignas(16) unsigned short Pbuf[8 * 16 * PROW];
    __shared__ alignas(16) float          rowScale[8 * 16];
    __shared__ alignas(16) float          rowL[8 * 16];
    __shared__ alignas(16) unsigned short Kst[2][32 * KROW];  // theta chunk
    __shared__ alignas(16) unsigned short Gst[2][64 * GROW];  // g chunk

    const int tid  = threadIdx.x;
    const int lane = tid & 31;
    const int wid  = tid >> 5;
    const int b    = blockIdx.x >> 5;                       // 32 blocks/batch
    const int n0   = (((blockIdx.x & 31) << 3) + wid) * 16;

    float*          Sw  = Sbuf + wid * 16 * SROW;
    unsigned short* Pw  = Pbuf + wid * 16 * PROW;
    float*          scW = rowScale + wid * 16;
    float*          lW  = rowL + wid * 16;

    const int h   = lane >> 4;   // half-wave select
    const int r16 = lane & 15;

    const unsigned short* thB = thetaT + (size_t)b * N_ * C8_;
    const unsigned short* gB  = gT + (size_t)b * C8_ * N_;

    // ---- Q fragments: A-layout, K=64 -> two K=32 fragments ----
    const unsigned short* phiB = phi + ((size_t)b * N_ + n0) * C8_;
    v16us aQ[2];
#pragma unroll
    for (int ks = 0; ks < 2; ++ks) {
        const unsigned short* p = phiB + r16 * C8_ + ks * 32 + h * 16;
        v8us lo = *(const v8us*)(p);
        v8us hi = *(const v8us*)(p + 8);
        aQ[ks] = __builtin_shufflevector(lo, hi, 0,1,2,3,4,5,6,7,8,9,10,11,12,13,14,15);
    }

    v8f yacc[4];
#pragma unroll
    for (int t = 0; t < 4; ++t) yacc[t] = (v8f)(0.f);

    float m_r = -1e30f, l_r = 0.f;   // owned by lanes 0..15 (one row each)

    // cooperative async stage of one 32-column chunk (theta 4KB + g 4KB)
    const int ktRow = tid >> 3, ktCol = (tid & 7) * 8;   // 16B per thread
    const int gtRow = tid >> 2, gtCol = (tid & 3) * 8;
    auto stage = [&](int buf, int m0) {
        async_load_b128(lds_off(&Kst[buf][ktRow * KROW + ktCol]),
                        thB + (size_t)(m0 + ktRow) * C8_ + ktCol);
        async_load_b128(lds_off(&Gst[buf][gtRow * GROW + gtCol]),
                        gB + (size_t)gtRow * N_ + m0 + gtCol);
    };

    stage(0, 0);

    for (int m0 = 0; m0 < N_; m0 += 32) {
        const int cur = (m0 >> 5) & 1;
        asm volatile("s_wait_asynccnt 0x0" ::: "memory");
        __syncthreads();                       // chunk `cur` resident in LDS
        if (m0 + 32 < N_) stage(cur ^ 1, m0 + 32);

        // ---- scores: 16x32 tile, two 16x16 subtiles, K=64 in two steps ----
        v8f s[2];
#pragma unroll
        for (int j = 0; j < 2; ++j) {
            s[j] = (v8f)(0.f);
#pragma unroll
            for (int ks = 0; ks < 2; ++ks) {
                v16us bk = *(const v16us*)&Kst[cur][(j * 16 + r16) * KROW + ks * 32 + h * 16];
                s[j] = wmma_bf16(aQ[ks], bk, s[j]);
            }
        }
        // spill scores to per-wave LDS tile (lane=col, vgpr i -> row h*8+i)
#pragma unroll
        for (int j = 0; j < 2; ++j)
#pragma unroll
            for (int i = 0; i < 8; ++i)
                Sw[(h * 8 + i) * SROW + j * 16 + r16] = s[j][i];
        asm volatile("s_wait_dscnt 0x0" ::: "memory");

        // ---- online softmax: lanes 0..15 each own one row ----
        if (lane < 16) {
            const float* srow = &Sw[lane * SROW];
            float cmax = srow[0];
#pragma unroll
            for (int j = 1; j < 32; ++j) cmax = fmaxf(cmax, srow[j]);
            float newm  = fmaxf(m_r, cmax);
            float scale = __expf(m_r - newm);
            float csum  = 0.f;
            unsigned short* prow = &Pw[lane * PROW];
#pragma unroll
            for (int j = 0; j < 32; ++j) {
                float p = __expf(srow[j] - newm);
                csum += p;
                prow[j] = f2bf(p);
            }
            l_r = l_r * scale + csum;
            m_r = newm;
            scW[lane] = scale;
        }
        asm volatile("s_wait_dscnt 0x0" ::: "memory");

        // ---- rescale running y accumulators ----
        float4 sa = *(const float4*)&scW[h * 8];
        float4 sb = *(const float4*)&scW[h * 8 + 4];
        float sc[8] = {sa.x, sa.y, sa.z, sa.w, sb.x, sb.y, sb.z, sb.w};
#pragma unroll
        for (int t = 0; t < 4; ++t)
#pragma unroll
            for (int i = 0; i < 8; ++i) yacc[t][i] *= sc[i];

        // ---- P as A-fragment (16x32) from LDS ----
        v8us plo = *(const v8us*)&Pw[r16 * PROW + h * 16];
        v8us phh = *(const v8us*)&Pw[r16 * PROW + h * 16 + 8];
        v16us aP = __builtin_shufflevector(plo, phh, 0,1,2,3,4,5,6,7,8,9,10,11,12,13,14,15);

        // ---- y += P * g : 4 c8-subtiles of 16 ----
#pragma unroll
        for (int t = 0; t < 4; ++t) {
            v16us bg = *(const v16us*)&Gst[cur][(t * 16 + r16) * GROW + h * 16];
            yacc[t] = wmma_bf16(aP, bg, yacc[t]);
        }
    }

    // ---- finalize: divide by row sums, store y (fp32, [B,N,C8] flat) ----
    if (lane < 16) lW[lane] = l_r;
    asm volatile("s_wait_dscnt 0x0" ::: "memory");
    float4 la = *(const float4*)&lW[h * 8];
    float4 lb = *(const float4*)&lW[h * 8 + 4];
    float li[8] = {la.x, la.y, la.z, la.w, lb.x, lb.y, lb.z, lb.w};
    float* yB = yflat + (size_t)b * N_ * C8_;
#pragma unroll
    for (int t = 0; t < 4; ++t)
#pragma unroll
        for (int i = 0; i < 8; ++i)
            yB[(size_t)(n0 + h * 8 + i) * C8_ + t * 16 + r16] = yacc[t][i] / li[i];
}

// ---------------------------------------------------------------------------
// Kernel 3a: per-(b,c) stats of x0 and of W_y (computed on the fly — AdaIN
// only needs W_y's mean/var, never the tensor).  .view quirk: channel k at
// pixel n is yflat[k*4096 + n].
// ---------------------------------------------------------------------------
__global__ void __launch_bounds__(256) stats_kernel(
    const float* __restrict__ x0, const float* __restrict__ yflat,
    const float* __restrict__ W_w, const float* __restrict__ W_b,
    float* __restrict__ stats)   // [B*C][4] = {mu_c, rstd_c, mu_s, std_s}
{
    __shared__ float red[256];
    const int c = blockIdx.x & (C_ - 1);
    const int b = blockIdx.x >> 9;
    const float* xs = x0 + ((size_t)b * C_ + c) * N_;
    const float* ys = yflat + (size_t)b * N_ * C8_;
    const float* wr = W_w + (size_t)c * C8_;
    const float  wb = W_b[c];

    float sx = 0.f, sxx = 0.f, sy = 0.f, syy = 0.f;
    for (int n = threadIdx.x; n < N_; n += 256) {
        float xv = xs[n];
        sx += xv; sxx += xv * xv;
        float wy = wb;
#pragma unroll
        for (int k = 0; k < C8_; ++k)
            wy = fmaf(wr[k], ys[(size_t)k * N_ + n], wy);
        sy += wy; syy += wy * wy;
    }
    float vals[4] = {sx, sxx, sy, syy};
    for (int q = 0; q < 4; ++q) {
        red[threadIdx.x] = vals[q];
        __syncthreads();
        for (int s = 128; s > 0; s >>= 1) {
            if (threadIdx.x < s) red[threadIdx.x] += red[threadIdx.x + s];
            __syncthreads();
        }
        vals[q] = red[0];
        __syncthreads();
    }
    if (threadIdx.x == 0) {
        float mx = vals[0] / N_, vx = vals[1] / N_ - mx * mx;
        float my = vals[2] / N_, vy = vals[3] / N_ - my * my;
        float* st = stats + (size_t)blockIdx.x * 4;
        st[0] = mx;
        st[1] = rsqrtf(vx + EPS_);
        st[2] = my;
        st[3] = sqrtf(vy + EPS_);
    }
}

// ---------------------------------------------------------------------------
// Kernel 3b: AdaIN elementwise: out = std_s * (x0 - mu_c) * rstd_c + mu_s
// ---------------------------------------------------------------------------
__global__ void __launch_bounds__(256) adain_kernel(
    const float* __restrict__ x0, const float* __restrict__ stats,
    float* __restrict__ out)
{
    size_t idx = (size_t)blockIdx.x * 256 + threadIdx.x;  // over B*C*N
    const int bc = (int)(idx >> 12);
    const float* st = stats + (size_t)bc * 4;
    out[idx] = st[3] * (x0[idx] - st[0]) * st[1] + st[2];
}

// ---------------------------------------------------------------------------
extern "C" void kernel_launch(void* const* d_in, const int* in_sizes, int n_in,
                              void* d_out, int out_size, void* d_ws, size_t ws_size,
                              hipStream_t stream)
{
    const float* x0   = (const float*)d_in[0];
    const float* x1   = (const float*)d_in[1];
    const float* g_w  = (const float*)d_in[2];
    const float* g_b  = (const float*)d_in[3];
    const float* th_w = (const float*)d_in[4];
    const float* th_b = (const float*)d_in[5];
    const float* ph_w = (const float*)d_in[6];
    const float* ph_b = (const float*)d_in[7];
    const float* W_w  = (const float*)d_in[8];
    const float* W_b  = (const float*)d_in[9];

    char* ws = (char*)d_ws;
    unsigned short* phi    = (unsigned short*)(ws);                     // 2 MB
    unsigned short* thetaT = (unsigned short*)(ws + (2ull << 20));      // 2 MB
    unsigned short* gT     = (unsigned short*)(ws + (4ull << 20));      // 2 MB
    float*          yflat  = (float*)(ws + (6ull << 20));               // 4 MB
    float*          stats  = (float*)(ws + (10ull << 20));              // 32 KB

    dim3 g1(N_ / 256, 3, B_);
    proj_kernel<<<g1, 256, 0, stream>>>(x0, x1, g_w, g_b, th_w, th_b,
                                        ph_w, ph_b, phi, thetaT, gT);
    attn_kernel<<<dim3(B_ * (N_ / 16) / 8), 256, 0, stream>>>(phi, thetaT, gT, yflat);
    stats_kernel<<<dim3(B_ * C_), 256, 0, stream>>>(x0, yflat, W_w, W_b, stats);
    adain_kernel<<<dim3((B_ * C_ * N_) / 256), 256, 0, stream>>>(x0, stats, (float*)d_out);
}
